// SeqClassLSTM_24842090840277
// MI455X (gfx1250) — matrix operand order repaired and checked by reading, not compiled, not run on hardware
//
#include <hip/hip_runtime.h>

typedef _Float16 v16h   __attribute__((ext_vector_type(16)));
typedef float    v8f    __attribute__((ext_vector_type(8)));
typedef _Float16 h2     __attribute__((ext_vector_type(2)));
typedef __fp16   fp16x2 __attribute__((ext_vector_type(2)));
typedef unsigned int uint32;

#define WAVES   4
#define XSTRIDE 132   // 128 + 4 floats pad -> lanes hit distinct LDS banks

// ---- fast transcendentals: gfx1250 has v_tanh_f32 (confirmed in round-2 asm) ----
#if __has_builtin(__builtin_amdgcn_tanhf)
__device__ __forceinline__ float ftanh(float x) { return __builtin_amdgcn_tanhf(x); }
#elif __has_builtin(__builtin_amdgcn_exp2f) && __has_builtin(__builtin_amdgcn_rcpf)
__device__ __forceinline__ float ftanh(float x) {
    float e = __builtin_amdgcn_exp2f(x * 2.885390081777927f);
    return 1.0f - 2.0f * __builtin_amdgcn_rcpf(e + 1.0f);
}
#else
__device__ __forceinline__ float ftanh(float x) { return tanhf(x); }
#endif

// r = xs * f16lo(wb) + f16hi(wb), without letting LICM expand wb to f32 regs.
__device__ __forceinline__ float fma_mix_lohi(float xs, uint32 wb) {
    float r;
    asm("v_fma_mix_f32 %0, %1, %2, %3 op_sel:[0,0,1] op_sel_hi:[0,1,1]"
        : "=v"(r) : "v"(xs), "v"(wb), "v"(wb));
    return r;
}

// One wave = 16 samples. gates^T = W_hh * h^T via v_wmma_f32_16x16x32_f16.
//  A tile t (16x32 f16): rows = gate 16t..16t+15, cols = hidden k (constant, regs)
//                        tiles 0-3,6,7 (sigmoid gates) pre-scaled by 0.5
//  B       (32x16 f16): rows = hidden k, cols = sample s (rebuilt per step in regs)
//  C/D tile (16x16 f32): lane L -> sample s=L%16, vgpr v -> gate row v+8*(L/16)
// Per lane: 16 hidden units of ONE sample: j = 16p + v + 8*hi.
__global__ __launch_bounds__(32 * WAVES)
void SeqClassLSTM_gfx1250_kernel(const float* __restrict__ x,
                                 const float* __restrict__ W_ih,
                                 const float* __restrict__ W_hh,
                                 const float* __restrict__ b_ih,
                                 const float* __restrict__ b_hh,
                                 const float* __restrict__ W_fc,
                                 const float* __restrict__ b_fc,
                                 float* __restrict__ out, int Btot)
{
    __shared__ float xlds[WAVES * 16 * XSTRIDE];

    const int lane = threadIdx.x & 31;
    const int wave = threadIdx.x >> 5;
    const int hi   = lane >> 4;
    const int s    = lane & 15;
    const int base = (blockIdx.x * WAVES + wave) * 16;
    if (base >= Btot) return;

    // ---------- stage x[base..base+15][0..127] into LDS (coalesced b128) ----------
    {
        const float4* gx = (const float4*)(x + (size_t)base * 128);
        float* wdst = xlds + wave * (16 * XSTRIDE);
        #pragma unroll
        for (int i = 0; i < 16; ++i) {
            int flat = (i * 32 + lane) * 4;
            float4 v = gx[i * 32 + lane];
            int sl = flat >> 7, tt = flat & 127;
            float* d = wdst + sl * XSTRIDE + tt;
            d[0] = v.x; d[1] = v.y; d[2] = v.z; d[3] = v.w;
        }
    }
    __syncthreads();

    // ---------- A tiles: W_hh as f16 (0.5x for sigmoid gates) ----------
    v16h a[8];
    #pragma unroll
    for (int t = 0; t < 8; ++t) {
        const float sc = (t == 4 || t == 5) ? 1.0f : 0.5f;   // tanh-gate tiles unscaled
        const float* row = W_hh + (size_t)(t * 16 + s) * 32;
        const int k0 = 16 * hi;
        #pragma unroll
        for (int e = 0; e < 8; ++e) {
            a[t][e]     = (_Float16)(sc * row[k0 + e]);
            a[t][e + 8] = (_Float16)(sc * row[k0 + 8 + e]);
        }
    }

    // ---------- per-lane packed (W_ih[g], b_ih[g]+b_hh[g]) f16 (pre-scaled) ----------
    uint32 wbp[64];
    #pragma unroll
    for (int t = 0; t < 8; ++t) {
        const float sc = (t == 4 || t == 5) ? 1.0f : 0.5f;
        #pragma unroll
        for (int v = 0; v < 8; ++v) {
            int g = t * 16 + 8 * hi + v;
            fp16x2 p = __builtin_amdgcn_cvt_pkrtz(sc * W_ih[g], sc * (b_ih[g] + b_hh[g]));
            wbp[t * 8 + v] = __builtin_bit_cast(uint32, p);
        }
    }

    float  cst[16];   // cell state, unit j = 16p + v + 8*hi
    uint32 bu[8];     // B operand (h^T as f16): lane half hi holds j = 16*hi + (0..15)
    #pragma unroll
    for (int i = 0; i < 16; ++i) cst[i] = 0.0f;
    #pragma unroll
    for (int i = 0; i < 8; ++i) bu[i] = 0u;

    const float* xrow = xlds + wave * (16 * XSTRIDE) + s * XSTRIDE;

    float4 xq = *(const float4*)(xrow);          // prefetch group 0
    for (int t4 = 0; t4 < 32; ++t4) {            // 128 timesteps, 4 per LDS b128 read
        float xv[4] = {xq.x, xq.y, xq.z, xq.w};
        xq = *(const float4*)(xrow + (((t4 + 1) & 31) * 4));   // branchless prefetch
        #pragma unroll
        for (int sub = 0; sub < 4; ++sub) {
            const float xs = xv[sub];

            union { v16h v; uint32 u[8]; } bun;
            #pragma unroll
            for (int i = 0; i < 8; ++i) bun.u[i] = bu[i];

            // 8 WMMAs: D = (sc*W_hh_tile) * h^T + sc*(x*W_ih + bias)
            v8f g8[8];
            #pragma unroll
            for (int t = 0; t < 8; ++t) {
                v8f c0;
                #pragma unroll
                for (int v = 0; v < 8; ++v)
                    c0[v] = fma_mix_lohi(xs, wbp[t * 8 + v]);
                g8[t] = __builtin_amdgcn_wmma_f32_16x16x32_f16(
                            false, a[t], false, bun.v, (short)0, c0, false, false);
            }

            // LSTM cell; i,f,o pre-scaled: sig = fma(0.5, tanh(z'), 0.5)
            uint32 pk0[4], pk1[4];
            #pragma unroll
            for (int p = 0; p < 2; ++p) {
                float hv[8];
                #pragma unroll
                for (int v = 0; v < 8; ++v) {
                    float si = fmaf(0.5f, ftanh(g8[0 + p][v]), 0.5f);
                    float sf = fmaf(0.5f, ftanh(g8[2 + p][v]), 0.5f);
                    float tg = ftanh(g8[4 + p][v]);
                    float so = fmaf(0.5f, ftanh(g8[6 + p][v]), 0.5f);
                    float cn = fmaf(sf, cst[p * 8 + v], si * tg);
                    cst[p * 8 + v] = cn;
                    hv[v] = so * ftanh(cn);
                }
                uint32* pk = p ? pk1 : pk0;
                #pragma unroll
                for (int q = 0; q < 4; ++q) {
                    fp16x2 ph = __builtin_amdgcn_cvt_pkrtz(hv[2 * q], hv[2 * q + 1]);
                    pk[q] = __builtin_bit_cast(uint32, ph);
                }
            }

            // in-register transpose into next step's B layout
            #pragma unroll
            for (int q = 0; q < 4; ++q) {
                uint32 send = hi ? pk0[q] : pk1[q];
                uint32 recv = (uint32)__shfl_xor((int)send, 16, 32);
                bu[q]     = hi ? recv   : pk0[q];
                bu[4 + q] = hi ? pk1[q] : recv;
            }
        }
    }

    // ---------- final FC straight from bu (f16 h): lane half hi holds j=16*hi+e ----------
    float o0 = 0.0f, o1 = 0.0f;
    {
        const float* w0 = W_fc + 0 * 32 + 16 * hi;
        const float* w1 = W_fc + 1 * 32 + 16 * hi;
        #pragma unroll
        for (int r = 0; r < 8; ++r) {
            h2 hp = __builtin_bit_cast(h2, bu[r]);
            float ha = (float)hp.x, hb = (float)hp.y;
            o0 = fmaf(ha, w0[2 * r], fmaf(hb, w0[2 * r + 1], o0));
            o1 = fmaf(ha, w1[2 * r], fmaf(hb, w1[2 * r + 1], o1));
        }
    }
    o0 += __shfl_xor(o0, 16, 32);
    o1 += __shfl_xor(o1, 16, 32);
    if (hi == 0) {
        float2 r; r.x = o0 + b_fc[0]; r.y = o1 + b_fc[1];
        *(float2*)(out + (size_t)(base + s) * 2) = r;
    }
}

extern "C" void kernel_launch(void* const* d_in, const int* in_sizes, int n_in,
                              void* d_out, int out_size, void* d_ws, size_t ws_size,
                              hipStream_t stream) {
    const float* x    = (const float*)d_in[0];
    const float* W_ih = (const float*)d_in[1];
    const float* W_hh = (const float*)d_in[2];
    const float* b_ih = (const float*)d_in[3];
    const float* b_hh = (const float*)d_in[4];
    const float* W_fc = (const float*)d_in[5];
    const float* b_fc = (const float*)d_in[6];

    const int T    = 128;
    const int Btot = in_sizes[0] / T;                 // x is [B, T, 1]
    const int samplesPerBlock = 16 * WAVES;
    const int blocks = (Btot + samplesPerBlock - 1) / samplesPerBlock;

    SeqClassLSTM_gfx1250_kernel<<<blocks, 32 * WAVES, 0, stream>>>(
        x, W_ih, W_hh, b_ih, b_hh, W_fc, b_fc, (float*)d_out, Btot);
}
